// CryptoAggregator_29317446762861
// MI455X (gfx1250) — matrix operand order repaired and checked by reading, not compiled, not run on hardware
//
#include <hip/hip_runtime.h>
#include <hip/hip_bf16.h>

typedef __attribute__((ext_vector_type(16))) _Float16 v16h;
typedef __attribute__((ext_vector_type(8)))  float    v8f;
typedef __attribute__((ext_vector_type(4)))  unsigned int v4u;
typedef __attribute__((ext_vector_type(8)))  int      v8i;
typedef __attribute__((ext_vector_type(4)))  int      v4i;

#define N_NODES 50000
#define N_EDGES 800000
#define DIM     256
#define DH      128

// d_ws byte offsets
#define WS_DEN_OFF  0                         // 50000 f32  (200000 B)
#define WS_W_OFF    200704                    // 800000 f32 (3.2 MB)
#define WS_W1S_OFF  (200704 + 3200000)        // 32768 f16  (64 KB), 32B aligned
// total ~3.47 MB of workspace

// ---------------------------------------------------------------------------
// 1) zero the num accumulator (d_out) and den
__global__ __launch_bounds__(256) void ca_init_kernel(float* __restrict__ out,
                                                      float* __restrict__ den) {
    size_t stride = (size_t)gridDim.x * blockDim.x;
    size_t i0 = (size_t)blockIdx.x * blockDim.x + threadIdx.x;
    const size_t total = (size_t)N_NODES * DIM;
    for (size_t i = i0; i < total; i += stride) out[i] = 0.0f;
    for (size_t i = i0; i < N_NODES; i += stride) den[i] = 0.0f;
}

// ---------------------------------------------------------------------------
// 2) repack W1 (f32 [256,128] row-major) into f16 WMMA B-fragment order:
//    frag(kt, nt)[lane][j]  where  K = 32*kt + (j<8 ? 8*(lane>>4)+j : 16+8*(lane>>4)+(j-8)),
//                                  N = 16*nt + (lane&15)
__global__ __launch_bounds__(256) void ca_swizzle_w1_kernel(const float* __restrict__ W1,
                                                            _Float16* __restrict__ w1s) {
    int i = blockIdx.x * blockDim.x + threadIdx.x;      // 32768 elements
    if (i >= 8 * 8 * 32 * 16) return;
    int j    = i & 15;
    int lane = (i >> 4) & 31;
    int nt   = (i >> 9) & 7;
    int kt   = i >> 12;
    int k = 32 * kt + (j < 8 ? 8 * (lane >> 4) + j : 16 + 8 * (lane >> 4) + (j - 8));
    int n = 16 * nt + (lane & 15);
    w1s[i] = (_Float16)W1[k * DH + n];
}

// ---------------------------------------------------------------------------
// TDM: DMA the 64KB of swizzled W1 fragments into LDS (one instruction).
// D# packed per CDNA5 ISA ch.8: 2D tensor, data_size=8B, dim0=8192 units.
__device__ inline void tdm_load_w1_to_lds(const _Float16* w1s, unsigned int lds_off) {
    unsigned long long ga = (unsigned long long)(const void*)w1s;
    v4u g0;
    g0[0] = 1u;                                        // count=1, user mode
    g0[1] = lds_off;                                   // lds_addr (bytes)
    g0[2] = (unsigned int)(ga & 0xFFFFFFFFu);          // global_addr[31:0]
    g0[3] = (unsigned int)((ga >> 32) & 0x01FFFFFFu)   // global_addr[56:32]
          | 0x80000000u;                               // type=2 ("image")
    v8i g1;
    g1[0] = (int)0x00030000;   // wg_mask=0, data_size=3 (8B), no pad/iter/barrier
    g1[1] = (int)0x20000000;   // tensor_dim0[15:0]=8192 in [31:16]
    g1[2] = (int)0x00010000;   // tensor_dim0[31:16]=0 ; tensor_dim1[15:0]=1
    g1[3] = (int)0x20000000;   // tensor_dim1[31:16]=0 ; tile_dim0=8192
    g1[4] = (int)0x00000001;   // tile_dim1=1, tile_dim2=0
    g1[5] = (int)8192;         // tensor_dim0_stride[31:0]
    g1[6] = (int)0x20000000;   // td0_stride[47:32]=0 ; td1_stride[15:0]=8192
    g1[7] = 0;                 // td1_stride[47:16]=0
    v4i gz = {0, 0, 0, 0};
#if defined(__clang_major__) && (__clang_major__ >= 23)
    v8i gz8 = {0, 0, 0, 0, 0, 0, 0, 0};
    __builtin_amdgcn_tensor_load_to_lds(g0, g1, gz, gz, gz8, 0);
#else
    __builtin_amdgcn_tensor_load_to_lds(g0, g1, gz, gz, 0);
#endif
    __builtin_amdgcn_s_wait_tensorcnt(0);
}

// ---------------------------------------------------------------------------
// 3) attention weights: one wave32 per 16-edge tile; 64 v_wmma per tile,
//    B fragments staged in LDS by the Tensor Data Mover.
__global__ __launch_bounds__(128) void ca_attn_kernel(const float* __restrict__ x,
                                                      const int*   __restrict__ ei,
                                                      const float* __restrict__ b1,
                                                      const float* __restrict__ W2,
                                                      const float* __restrict__ b2,
                                                      const _Float16* __restrict__ w1s,
                                                      float* __restrict__ wbuf) {
    __shared__ _Float16 s_w1[8 * 8 * 32 * 16];         // 64 KB of B fragments

    const int wave  = blockIdx.x * 4 + (threadIdx.x >> 5);   // tile id, exactly 50000
    const int lane  = threadIdx.x & 31;
    const int g     = lane >> 4;        // K-half selector for A/B fragments
    const int nsub  = lane & 15;        // row-in-tile for A gather, N for B/C
    const int ebase = wave * 16;

    // wave 0 DMAs W1 fragments into LDS; everyone meets at the barrier
    if ((threadIdx.x >> 5) == 0) {
        tdm_load_w1_to_lds(w1s, (unsigned int)(size_t)(&s_w1[0]));
    }

    // gather pointer for this lane's edge row (lanes l and l+16 share a row)
    const int col = ei[N_EDGES + ebase + nsub];
    const float* rowp = x + (size_t)col * DIM;
    __builtin_prefetch(rowp, 0, 3);     // global_prefetch into cache hierarchy

    // --- load all 8 A fragments (16x32 f16 each) straight from the gather ---
    // lane l (row m=l&15, half g): f16 elem j -> K = 32*kt + (j<8 ? 8g+j : 16+8g+(j-8))
    v16h afrag[8];
#pragma unroll
    for (int kt = 0; kt < 8; ++kt) {
        const float4* f4a = (const float4*)(rowp + kt * 32 + 8 * g);
        const float4* f4b = (const float4*)(rowp + kt * 32 + 16 + 8 * g);
        float4 r0 = f4a[0], r1 = f4a[1];
        float4 r2 = f4b[0], r3 = f4b[1];
        v16h a;
        a[0] = (_Float16)r0.x;  a[1] = (_Float16)r0.y;
        a[2] = (_Float16)r0.z;  a[3] = (_Float16)r0.w;
        a[4] = (_Float16)r1.x;  a[5] = (_Float16)r1.y;
        a[6] = (_Float16)r1.z;  a[7] = (_Float16)r1.w;
        a[8] = (_Float16)r2.x;  a[9] = (_Float16)r2.y;
        a[10] = (_Float16)r2.z; a[11] = (_Float16)r2.w;
        a[12] = (_Float16)r3.x; a[13] = (_Float16)r3.y;
        a[14] = (_Float16)r3.z; a[15] = (_Float16)r3.w;
        afrag[kt] = a;
    }

    __syncthreads();                    // LDS W1 fragments now visible

    // --- 8 N-tiles x 8 K-tiles of v_wmma_f32_16x16x32_f16; fold ReLU+W2 dot ---
    const v16h* bfr = (const v16h*)s_w1;       // fragment index (kt*8+nt)*32+lane
    float p[8] = {0.f, 0.f, 0.f, 0.f, 0.f, 0.f, 0.f, 0.f};
#pragma unroll
    for (int nt = 0; nt < 8; ++nt) {
        v8f acc = {0.f, 0.f, 0.f, 0.f, 0.f, 0.f, 0.f, 0.f};
#pragma unroll
        for (int kt = 0; kt < 8; ++kt) {
            v16h b = bfr[(kt * 8 + nt) * 32 + lane];
            acc = __builtin_amdgcn_wmma_f32_16x16x32_f16(
                /*neg_a=*/false, afrag[kt], /*neg_b=*/false, b,
                /*c_mod=*/(short)0, acc, /*reuse_a=*/false, /*reuse_b=*/false);
        }
        const float b1v = b1[16 * nt + nsub];
        const float w2v = W2[16 * nt + nsub];
        // C/D layout: lane holds N=nsub, M = 8*g + r in VGPR r
#pragma unroll
        for (int r = 0; r < 8; ++r) {
            float hv = acc[r] + b1v;
            hv = hv > 0.f ? hv : 0.f;          // ReLU
            p[r] += hv * w2v;                  // partial dot with W2
        }
    }

    // reduce the N dimension across the 16 lanes of each half-wave
#pragma unroll
    for (int r = 0; r < 8; ++r) {
        float v = p[r];
        v += __shfl_xor(v, 1, 16);
        v += __shfl_xor(v, 2, 16);
        v += __shfl_xor(v, 4, 16);
        v += __shfl_xor(v, 8, 16);
        p[r] = v;
    }

    const float b2v = b2[0];
    if (nsub == 0) {                            // lane 0 -> m=0..7, lane 16 -> m=8..15
#pragma unroll
        for (int r = 0; r < 8; ++r) {
            float s = p[r] + b2v;
            wbuf[ebase + 8 * g + r] = 1.0f / (1.0f + __expf(-s));   // sigmoid
        }
    }
}

// ---------------------------------------------------------------------------
// 4) weighted scatter: one wave per edge, 8 floats per lane
__global__ __launch_bounds__(256) void ca_scatter_kernel(const float* __restrict__ x,
                                                         const int*   __restrict__ ei,
                                                         const float* __restrict__ wbuf,
                                                         float* __restrict__ out,
                                                         float* __restrict__ den) {
    const int e = blockIdx.x * 8 + (threadIdx.x >> 5);
    if (e >= N_EDGES) return;
    const int lane = threadIdx.x & 31;
    const int row = ei[e];
    const int col = ei[N_EDGES + e];
    const float wv = wbuf[e];

    const float4* src = (const float4*)(x + (size_t)col * DIM) + lane * 2;
    float4 v0 = src[0], v1 = src[1];
    float* dst = out + (size_t)row * DIM + lane * 8;
    atomicAdd(dst + 0, v0.x * wv);
    atomicAdd(dst + 1, v0.y * wv);
    atomicAdd(dst + 2, v0.z * wv);
    atomicAdd(dst + 3, v0.w * wv);
    atomicAdd(dst + 4, v1.x * wv);
    atomicAdd(dst + 5, v1.y * wv);
    atomicAdd(dst + 6, v1.z * wv);
    atomicAdd(dst + 7, v1.w * wv);
    if (lane == 0) atomicAdd(den + row, wv);
}

// ---------------------------------------------------------------------------
// 5) normalize: out = den>0 ? num/max(den,1e-12) : 0
__global__ __launch_bounds__(256) void ca_norm_kernel(float* __restrict__ out,
                                                      const float* __restrict__ den) {
    size_t i = (size_t)blockIdx.x * blockDim.x + threadIdx.x;
    if (i >= (size_t)N_NODES * DIM) return;
    float d = den[i >> 8];          // i / 256
    float v = out[i];
    out[i] = (d > 0.f) ? (v / fmaxf(d, 1e-12f)) : 0.f;
}

// ---------------------------------------------------------------------------
extern "C" void kernel_launch(void* const* d_in, const int* in_sizes, int n_in,
                              void* d_out, int out_size, void* d_ws, size_t ws_size,
                              hipStream_t stream) {
    const float* x  = (const float*)d_in[0];
    const int*   ei = (const int*)d_in[1];
    const float* W1 = (const float*)d_in[2];
    const float* b1 = (const float*)d_in[3];
    const float* W2 = (const float*)d_in[4];
    const float* b2 = (const float*)d_in[5];
    float* out = (float*)d_out;

    char* ws = (char*)d_ws;
    float*     den  = (float*)(ws + WS_DEN_OFF);
    float*     wbuf = (float*)(ws + WS_W_OFF);
    _Float16*  w1s  = (_Float16*)(ws + WS_W1S_OFF);

    ca_init_kernel<<<2048, 256, 0, stream>>>(out, den);
    ca_swizzle_w1_kernel<<<(8 * 8 * 32 * 16 + 255) / 256, 256, 0, stream>>>(W1, w1s);
    // 800000 edges / 16 per wave = 50000 waves; 4 waves per block
    ca_attn_kernel<<<12500, 128, 0, stream>>>(x, ei, b1, W2, b2, w1s, wbuf);
    // one wave per edge; 8 waves per block
    ca_scatter_kernel<<<(N_EDGES + 7) / 8, 256, 0, stream>>>(x, ei, wbuf, out, den);
    ca_norm_kernel<<<((size_t)N_NODES * DIM + 255) / 256, 256, 0, stream>>>(out, den);
}
